// TransformerEncoderLayer_18433999634873
// MI455X (gfx1250) — matrix operand order repaired and checked by reading, not compiled, not run on hardware
//
#include <hip/hip_runtime.h>
#include <hip/hip_bf16.h>

typedef __attribute__((ext_vector_type(16))) _Float16 v16h;
typedef __attribute__((ext_vector_type(8)))  _Float16 h8;
typedef __attribute__((ext_vector_type(8)))  float    v8f;

union AccU { v8f v; float f[8]; };

__device__ __forceinline__ h8 h8_zero() {
    union { h8 v; unsigned long long q[2]; } z;
    z.q[0] = 0ull; z.q[1] = 0ull;
    return z.v;
}

// ---------------------------------------------------------------------------
// Tensor Data Mover staging (gfx1250). 2D tile load, data_size=2B,
// LDS padding: 64B of data (16 DWORDs, pad_interval code 3) then 16B pad
// (4 DWORDs, pad_amount code 3)  ->  matches the 40-half LDS row pitch.
// ---------------------------------------------------------------------------
#define TDM_OK __has_builtin(__builtin_amdgcn_tensor_load_to_lds)

#if TDM_OK
typedef unsigned int v4u __attribute__((ext_vector_type(4)));
typedef int          v4i __attribute__((ext_vector_type(4)));
typedef int          v8i __attribute__((ext_vector_type(8)));

__device__ __forceinline__ void tdm_load_tile_2d(
    unsigned ldsAddr, const _Float16* gsrc,
    unsigned tileK, unsigned tileRows, unsigned long long rowStrideElems) {
    union { v4u v; unsigned w[4]; } g0;
    union { v8i v; unsigned w[8]; } g1;
    union { v4i v; unsigned w[4]; } gz;
    unsigned long long ga = (unsigned long long)gsrc;
    const unsigned dim0 = 1u << 20, dim1 = 1u << 20;  // generous: no OOB clip
    g0.w[0] = 1u;                                       // count=1, user D#
    g0.w[1] = ldsAddr;                                  // LDS byte address
    g0.w[2] = (unsigned)(ga & 0xFFFFFFFFu);             // global_addr lo
    g0.w[3] = (unsigned)((ga >> 32) & 0x1FFFFFFu) | (2u << 30);  // hi | type=2
    g1.w[0] = (1u << 16)    // data_size = 2 bytes
            | (1u << 20)    // pad_enable
            | (3u << 22)    // pad_interval = 16 DWORDs
            | (3u << 25);   // pad_amount  = 4 DWORDs
    g1.w[1] = (dim0 & 0xFFFFu) << 16;                   // tensor_dim0[15:0]
    g1.w[2] = (dim0 >> 16) | ((dim1 & 0xFFFFu) << 16);  // dim0 hi | dim1 lo
    g1.w[3] = (dim1 >> 16) | (tileK << 16);             // dim1 hi | tile_dim0
    g1.w[4] = tileRows;                                 // tile_dim1 (tile_dim2=0)
    g1.w[5] = (unsigned)(rowStrideElems & 0xFFFFFFFFull);
    g1.w[6] = (unsigned)((rowStrideElems >> 32) & 0xFFFFull);
    g1.w[7] = 0;
    gz.w[0] = gz.w[1] = gz.w[2] = gz.w[3] = 0;
#if __clang_major__ >= 23
    union { v8i v; unsigned w[8]; } gz8;
    for (int i = 0; i < 8; ++i) gz8.w[i] = 0;
    __builtin_amdgcn_tensor_load_to_lds(g0.v, g1.v, gz.v, gz.v, gz8.v, 0);
#else
    __builtin_amdgcn_tensor_load_to_lds(g0.v, g1.v, gz.v, gz.v, 0);
#endif
}
constexpr bool kUseTdm = true;
#else
constexpr bool kUseTdm = false;
#endif

// ---------------------------------------------------------------------------
// Fragment loads (ISA 7.12.2, wave32), 2x ds_load_b128 each.
// ---------------------------------------------------------------------------
__device__ __forceinline__ v16h load_a_frag(const _Float16 (*s)[40], int rowBase, int lane) {
    int row = rowBase + (lane & 15);
    int hi  = (lane >> 4) << 3;
    union { v16h v; h8 h[2]; } u;
    u.h[0] = *(const h8*)&s[row][hi];
    u.h[1] = *(const h8*)&s[row][16 + hi];
    return u.v;
}

__device__ __forceinline__ v16h load_b_frag(const _Float16 (*s)[40], int colBase, int lane) {
    int row = colBase + (lane & 15);
    int kb  = (lane >> 4) << 3;
    union { v16h v; h8 h[2]; } u;
    u.h[0] = *(const h8*)&s[row][kb];
    u.h[1] = *(const h8*)&s[row][16 + kb];
    return u.v;
}

// ---------------------------------------------------------------------------
// Spectral norm via power iteration: writes 1/sigma_max(W) for W[O][I]
// ---------------------------------------------------------------------------
__global__ __launch_bounds__(256) void power_iter(const float* __restrict__ W,
                                                  int O, int I,
                                                  float* __restrict__ sig_inv) {
    __shared__ float u[1024];
    __shared__ float v[1024];
    __shared__ float red[256];
    const int t = threadIdx.x;
    for (int i = t; i < O; i += 256) u[i] = 1.0f;
    __syncthreads();
    float sigma = 1.0f;
    for (int it = 0; it < 8; ++it) {
        for (int j = t; j < I; j += 256) {
            float s = 0.f;
            for (int o = 0; o < O; ++o) s += W[(long)o * I + j] * u[o];
            v[j] = s;
        }
        __syncthreads();
        float ls = 0.f;
        for (int j = t; j < I; j += 256) ls += v[j] * v[j];
        red[t] = ls; __syncthreads();
        for (int s = 128; s > 0; s >>= 1) { if (t < s) red[t] += red[t + s]; __syncthreads(); }
        float nv = sqrtf(red[0]) + 1e-12f;
        __syncthreads();
        for (int j = t; j < I; j += 256) v[j] /= nv;
        __syncthreads();
        for (int o = t; o < O; o += 256) {
            float s = 0.f;
            for (int j = 0; j < I; ++j) s += W[(long)o * I + j] * v[j];
            u[o] = s;
        }
        __syncthreads();
        float lu = 0.f;
        for (int o = t; o < O; o += 256) lu += u[o] * u[o];
        red[t] = lu; __syncthreads();
        for (int s = 128; s > 0; s >>= 1) { if (t < s) red[t] += red[t + s]; __syncthreads(); }
        sigma = sqrtf(red[0]);
        __syncthreads();
        float inv = 1.0f / (sigma + 1e-12f);
        for (int o = t; o < O; o += 256) u[o] *= inv;
        __syncthreads();
    }
    if (t == 0) sig_inv[0] = 1.0f / (sigma + 1e-12f);
}

__global__ __launch_bounds__(256) void scale_convert(const float* __restrict__ in,
                                                     _Float16* __restrict__ out,
                                                     long n, const float* __restrict__ scale) {
    float s = scale ? scale[0] : 1.0f;
    for (long i = (long)blockIdx.x * 256 + threadIdx.x; i < n; i += (long)gridDim.x * 256)
        out[i] = (_Float16)(in[i] * s);
}

// Repack conv weights [OC][IC][3] f32 -> [3][OC][IC] f16 (k-contiguous per tap)
__global__ __launch_bounds__(256) void repack_conv(const float* __restrict__ w,
                                                   _Float16* __restrict__ out,
                                                   int OC, int IC) {
    long plane = (long)OC * IC;
    long n = plane * 3;
    for (long i = (long)blockIdx.x * 256 + threadIdx.x; i < n; i += (long)gridDim.x * 256) {
        long tap = i / plane;
        long rem = i - tap * plane;
        long oc = rem / IC, ic = rem - oc * IC;
        out[i] = (_Float16)w[(oc * IC + ic) * 3 + tap];
    }
}

// LDS-tiled transpose + convert: in f32 [R][Cc] -> out f16 [Cc][R]  (batched)
__global__ __launch_bounds__(256) void transpose_f32_to_f16(
    const float* __restrict__ in, _Float16* __restrict__ out,
    int R, int Cc, long strideIn, long strideOut) {
    __shared__ float tile[32][33];
    const int bx = blockIdx.x * 32;
    const int by = blockIdx.y * 32;
    const long bi = (long)blockIdx.z * strideIn;
    const long bo = (long)blockIdx.z * strideOut;
    const int tx = threadIdx.x & 31;
    const int ty = threadIdx.x >> 5;
    for (int i = ty; i < 32; i += 8)
        tile[i][tx] = in[bi + (long)(by + i) * Cc + (bx + tx)];
    __syncthreads();
    for (int i = ty; i < 32; i += 8)
        out[bo + (long)(bx + i) * R + (by + tx)] = (_Float16)tile[tx][i];
}

// ---------------------------------------------------------------------------
// Unified batched WMMA GEMM / conv1d(K=3,'same'), double-buffered pipeline.
//   acc[M][N] = sum_tap  A[tap](MxK) * Bt^T     (f16 in, f32 acc)
//   USE_TDM: stage tiles with tensor_load_to_lds (wave 0) + s_wait_tensorcnt
//   else   : per-lane b128 loads issued before WMMA, ds_store after (pipelined)
//   EPI 0: v = acc                EPI 1: v = 2*res + gamma[0]*acc
//   EPI 2: v = relu(acc+bias)     EPI 3: v = relu(acc+bias) + res
// Block tile 64x128, BK=32, 256 threads = 8 waves (2x4), wave tile 32x32.
// ---------------------------------------------------------------------------
template <int NTAPS, int EPI, bool STORE_T, bool OUT16, bool USE_TDM>
__global__ __launch_bounds__(256) void wmma_gemm(
    const _Float16* __restrict__ A, const _Float16* __restrict__ Bt,
    void* __restrict__ Cout,
    int M, int N, int K, int ldb, int ldc,
    long strideA, long strideB, long strideC,
    const float* __restrict__ bias,
    const float* __restrict__ res, long strideRes,
    const float* __restrict__ gamma) {
    __shared__ _Float16 sA[2][64][40];
    __shared__ _Float16 sB[2][128][40];

    const int tid  = threadIdx.x;
    const int lane = tid & 31;
    const int wid  = tid >> 5;
    const int wm   = (wid >> 2) * 32;
    const int wn   = (wid & 3) * 32;
    const int mBase = blockIdx.y * 64;
    const int nBase = blockIdx.x * 128;
    const long aB = (long)blockIdx.z * strideA;
    const long bB = (long)blockIdx.z * strideB;
    const long cB = (long)blockIdx.z * strideC;

    AccU acc[2][2];
#pragma unroll
    for (int i = 0; i < 2; ++i)
#pragma unroll
        for (int j = 0; j < 2; ++j)
#pragma unroll
            for (int r = 0; r < 8; ++r) acc[i][j].f[r] = 0.0f;

    const int KS = K >> 5;          // k-steps per tap
    const int S  = NTAPS * KS;      // total pipeline steps
    const int ar = tid >> 2, ac8 = (tid & 3) << 3;   // sA: row / 8-half chunk
    const int br0 = tid >> 2, bc8 = (tid & 3) << 3;  // sB rows br0 and br0+64

    h8 ra, rb0, rb1;  // register staging (non-TDM path)

    auto loadRegs = [&](int s) {
        int tap = s / KS;
        int k0  = (s - tap * KS) << 5;
        int shift = (NTAPS == 3) ? (tap - 1) : 0;
        ra = *(const h8*)(A + aB + (long)tap * M * K + (long)(mBase + ar) * K + k0 + ac8);
        int row0 = nBase + br0 + shift;
        int row1 = row0 + 64;
        rb0 = (NTAPS == 3 && (row0 < 0 || row0 >= N))
                  ? h8_zero() : *(const h8*)(Bt + bB + (long)row0 * ldb + k0 + bc8);
        rb1 = (NTAPS == 3 && (row1 < 0 || row1 >= N))
                  ? h8_zero() : *(const h8*)(Bt + bB + (long)row1 * ldb + k0 + bc8);
    };
    auto storeRegs = [&](int b) {
        *(h8*)&sA[b][ar][ac8] = ra;
        *(h8*)&sB[b][br0][bc8] = rb0;
        *(h8*)&sB[b][br0 + 64][bc8] = rb1;
    };
    auto stageTdm = [&](int s, int b) {
#if TDM_OK
        int k0 = (s % KS) << 5;  // NTAPS==1 only
        if (wid == 0) {
            tdm_load_tile_2d((unsigned)(unsigned long long)&sA[b][0][0],
                             A + aB + (long)mBase * K + k0, 32u, 64u,
                             (unsigned long long)K);
            tdm_load_tile_2d((unsigned)(unsigned long long)&sB[b][0][0],
                             Bt + bB + (long)nBase * ldb + k0, 32u, 128u,
                             (unsigned long long)ldb);
        }
#endif
    };

    int buf = 0;
    if constexpr (USE_TDM) {
#if TDM_OK
        stageTdm(0, 0);
        if (wid == 0) __builtin_amdgcn_s_wait_tensorcnt(0);
#endif
        __syncthreads();
    } else {
        loadRegs(0);
        storeRegs(0);
        __syncthreads();
    }

    for (int s = 0; s < S; ++s) {
        const int nb = buf ^ 1;
        const bool more = (s + 1 < S);
        if constexpr (USE_TDM) {
            if (more) stageTdm(s + 1, nb);
        } else {
            if (more) loadRegs(s + 1);   // global loads issued before WMMA block
        }

        v16h af[2], bf[2];
#pragma unroll
        for (int i = 0; i < 2; ++i) af[i] = load_a_frag(sA[buf], wm + i * 16, lane);
#pragma unroll
        for (int j = 0; j < 2; ++j) bf[j] = load_b_frag(sB[buf], wn + j * 16, lane);
#pragma unroll
        for (int i = 0; i < 2; ++i)
#pragma unroll
            for (int j = 0; j < 2; ++j)
                acc[i][j].v = __builtin_amdgcn_wmma_f32_16x16x32_f16(
                    false, af[i], false, bf[j], (short)0, acc[i][j].v, false, false);

        if constexpr (USE_TDM) {
#if TDM_OK
            if (more && wid == 0) __builtin_amdgcn_s_wait_tensorcnt(0);
#endif
        } else {
            if (more) storeRegs(nb);     // LDS stores after WMMA (overlap)
        }
        __syncthreads();
        buf = nb;
    }

    _Float16* out16 = (_Float16*)Cout;
    float*    outf  = (float*)Cout;
    const float gam = (EPI == 1) ? gamma[0] : 0.0f;
    const int cm = (lane >> 4) << 3;
    const int cn = lane & 15;
#pragma unroll
    for (int i = 0; i < 2; ++i) {
#pragma unroll
        for (int j = 0; j < 2; ++j) {
            const int gm0 = mBase + wm + i * 16 + cm;
            const int gn  = nBase + wn + j * 16 + cn;
            union { h8 v; _Float16 e[8]; } ov;
#pragma unroll
            for (int r = 0; r < 8; ++r) {
                float v = acc[i][j].f[r];
                if (EPI == 1) {
                    long ro = (long)blockIdx.z * strideRes + (long)(gm0 + r) * N + gn;
                    v = 2.0f * res[ro] + gam * v;
                } else if (EPI == 2) {
                    v = fmaxf(v + bias[gm0 + r], 0.0f);
                } else if (EPI == 3) {
                    long ro = (long)blockIdx.z * strideRes + (long)(gm0 + r) * N + gn;
                    v = fmaxf(v + bias[gm0 + r], 0.0f) + res[ro];
                }
                if (STORE_T) {
                    ov.e[r] = (_Float16)v;
                } else if (OUT16) {
                    out16[cB + (long)(gm0 + r) * ldc + gn] = (_Float16)v;
                } else {
                    outf[cB + (long)(gm0 + r) * ldc + gn] = v;
                }
            }
            if (STORE_T)
                *(h8*)(out16 + cB + (long)gn * ldc + gm0) = ov.v;
        }
    }
}

// ---------------------------------------------------------------------------
// Row softmax over 512 f32 logits -> f16 attn ; one wave per row
// ---------------------------------------------------------------------------
__global__ __launch_bounds__(256) void softmax512(const float* __restrict__ logits,
                                                  _Float16* __restrict__ attn, int rows) {
    const int wid  = threadIdx.x >> 5;
    const int lane = threadIdx.x & 31;
    const long row = (long)blockIdx.x * 8 + wid;
    if (row >= rows) return;
    const float* p = logits + row * 512;
    _Float16*    q = attn + row * 512;
    float v[16];
    float m = -1e30f;
#pragma unroll
    for (int i = 0; i < 16; ++i) { v[i] = p[lane + i * 32]; m = fmaxf(m, v[i]); }
#pragma unroll
    for (int o = 16; o > 0; o >>= 1) m = fmaxf(m, __shfl_xor(m, o, 32));
    float s = 0.f;
#pragma unroll
    for (int i = 0; i < 16; ++i) { v[i] = __expf(v[i] - m); s += v[i]; }
#pragma unroll
    for (int o = 16; o > 0; o >>= 1) s += __shfl_xor(s, o, 32);
    float inv = 1.0f / s;
#pragma unroll
    for (int i = 0; i < 16; ++i) q[lane + i * 32] = (_Float16)(v[i] * inv);
}

// ---------------------------------------------------------------------------
// MaxPool2d(2,2) over (c,w). Input stored transposed: inT f16 [Win][Cin].
// ---------------------------------------------------------------------------
template <bool OUT_T>
__global__ __launch_bounds__(256) void maxpool_reshape(
    const _Float16* __restrict__ inT, _Float16* __restrict__ out,
    int Cin, int Win, long strideIn, long strideOut) {
    const long nOut = (long)(Cin >> 1) * (Win >> 1);
    const long bi = (long)blockIdx.z * strideIn;
    const long bo = (long)blockIdx.z * strideOut;
    const int Wh = Win >> 1, Wq = Win >> 2;
    for (long f = (long)blockIdx.x * 256 + threadIdx.x; f < nOut; f += (long)gridDim.x * 256) {
        long c2 = f / Wh, w2 = f % Wh;
        const _Float16* base = inT + bi;
        float a00 = (float)base[(2 * w2) * Cin + 2 * c2];
        float a01 = (float)base[(2 * w2 + 1) * Cin + 2 * c2];
        float a10 = (float)base[(2 * w2) * Cin + 2 * c2 + 1];
        float a11 = (float)base[(2 * w2 + 1) * Cin + 2 * c2 + 1];
        float mx = fmaxf(fmaxf(a00, a01), fmaxf(a10, a11));
        long c = f / Wq, vv = f % Wq;
        if (OUT_T) out[bo + vv * Cin + c] = (_Float16)mx;
        else       out[bo + c * Wq + vv] = (_Float16)mx;
    }
}

// ---------------------------------------------------------------------------
// Host-side orchestration
// ---------------------------------------------------------------------------
extern "C" void kernel_launch(void* const* d_in, const int* in_sizes, int n_in,
                              void* d_out, int out_size, void* d_ws, size_t ws_size,
                              hipStream_t stream) {
    const int B = 4, CH = 1024, W = 2048;
    const int C8 = CH / 8;   // 128
    const int C2 = CH / 2;   // 512
    const int W4 = W / 4;    // 512

    const float* src     = (const float*)d_in[0];
    const float* w_theta = (const float*)d_in[1];
    const float* w_phi   = (const float*)d_in[2];
    const float* w_g     = (const float*)d_in[3];
    const float* w_o     = (const float*)d_in[4];
    const float* gamma   = (const float*)d_in[5];
    const float* w_c1    = (const float*)d_in[6];
    const float* b_c1    = (const float*)d_in[7];
    const float* w_c2    = (const float*)d_in[8];
    const float* b_c2    = (const float*)d_in[9];
    const float* w_c3    = (const float*)d_in[10];
    const float* b_c3    = (const float*)d_in[11];
    const float* w_c4    = (const float*)d_in[12];
    const float* b_c4    = (const float*)d_in[13];

    char* base = (char*)d_ws;
    size_t cur = 0;
    auto alloc = [&](size_t bytes) -> char* {
        char* p = base + cur;
        cur = (cur + bytes + 255) & ~(size_t)255;
        return p;
    };
    float*    sig     = (float*)alloc(16 * 4);
    _Float16* wthetaH = (_Float16*)alloc((size_t)C8 * CH * 2);
    _Float16* wphiH   = (_Float16*)alloc((size_t)C8 * CH * 2);
    _Float16* wgH     = (_Float16*)alloc((size_t)C2 * CH * 2);
    _Float16* woH     = (_Float16*)alloc((size_t)CH * C2 * 2);
    _Float16* wc1p    = (_Float16*)alloc((size_t)3 * C2 * CH * 2);
    _Float16* wc2p    = (_Float16*)alloc((size_t)3 * C2 * C2 * 2);
    _Float16* wc3p    = (_Float16*)alloc((size_t)3 * C2 * C2 * 2);
    _Float16* wc4p    = (_Float16*)alloc((size_t)3 * CH * C2 * 2);
    _Float16* srcT    = (_Float16*)alloc((size_t)B * W * CH * 2);   // [W][CH]
    _Float16* thetaT  = (_Float16*)alloc((size_t)B * W * C8 * 2);   // [W][C8]
    _Float16* phiFT   = (_Float16*)alloc((size_t)B * W * C8 * 2);   // [W][C8]
    _Float16* phiPT   = (_Float16*)alloc((size_t)B * W4 * C8 * 2);  // [W4][C8]
    _Float16* gFT     = (_Float16*)alloc((size_t)B * W * C2 * 2);   // [W][C2]
    _Float16* gP      = (_Float16*)alloc((size_t)B * C2 * W4 * 2);  // [C2][W4]
    float*    logits  = (float*)alloc((size_t)B * W * W4 * 4);      // [W][W4] f32
    _Float16* attnH   = (_Float16*)alloc((size_t)B * W * W4 * 2);   // [W][W4]
    _Float16* attngT  = (_Float16*)alloc((size_t)B * W * C2 * 2);   // [W][C2]
    _Float16* xT  = (_Float16*)logits;  // logits dead after softmax (same bytes)
    _Float16* h1T = gFT;                // gFT dead after pool_g
    _Float16* h2T = attnH;              // attnH dead after attn_g GEMM
    _Float16* h3T = attngT;             // attngT dead after x GEMM
    float*    out = (float*)d_out;

    // 1) spectral norms + weight conversion / repack / src transpose
    power_iter<<<1, 256, 0, stream>>>(w_theta, C8, CH, sig + 0);
    power_iter<<<1, 256, 0, stream>>>(w_phi,   C8, CH, sig + 1);
    power_iter<<<1, 256, 0, stream>>>(w_g,     C2, CH, sig + 2);
    power_iter<<<1, 256, 0, stream>>>(w_o,     CH, C2, sig + 3);
    scale_convert<<<256, 256, 0, stream>>>(w_theta, wthetaH, (long)C8 * CH, sig + 0);
    scale_convert<<<256, 256, 0, stream>>>(w_phi,   wphiH,   (long)C8 * CH, sig + 1);
    scale_convert<<<512, 256, 0, stream>>>(w_g,     wgH,     (long)C2 * CH, sig + 2);
    scale_convert<<<512, 256, 0, stream>>>(w_o,     woH,     (long)CH * C2, sig + 3);
    repack_conv<<<1024, 256, 0, stream>>>(w_c1, wc1p, C2, CH);
    repack_conv<<<512,  256, 0, stream>>>(w_c2, wc2p, C2, C2);
    repack_conv<<<512,  256, 0, stream>>>(w_c3, wc3p, C2, C2);
    repack_conv<<<1024, 256, 0, stream>>>(w_c4, wc4p, CH, C2);
    transpose_f32_to_f16<<<dim3(W / 32, CH / 32, B), 256, 0, stream>>>(
        src, srcT, CH, W, (long)CH * W, (long)CH * W);

    // 2) projections (STORE_T -> [W][O] f16)
    wmma_gemm<1, 0, true, true, kUseTdm><<<dim3(W / 128, C8 / 64, B), 256, 0, stream>>>(
        wthetaH, srcT, thetaT, C8, W, CH, CH, C8,
        0, (long)W * CH, (long)W * C8, nullptr, nullptr, 0, nullptr);
    wmma_gemm<1, 0, true, true, kUseTdm><<<dim3(W / 128, C8 / 64, B), 256, 0, stream>>>(
        wphiH, srcT, phiFT, C8, W, CH, CH, C8,
        0, (long)W * CH, (long)W * C8, nullptr, nullptr, 0, nullptr);
    wmma_gemm<1, 0, true, true, kUseTdm><<<dim3(W / 128, C2 / 64, B), 256, 0, stream>>>(
        wgH, srcT, gFT, C2, W, CH, CH, C2,
        0, (long)W * CH, (long)W * C2, nullptr, nullptr, 0, nullptr);

    // 3) pool + reshape
    maxpool_reshape<true><<<dim3(256, 1, B), 256, 0, stream>>>(
        phiFT, phiPT, C8, W, (long)W * C8, (long)W4 * C8);
    maxpool_reshape<false><<<dim3(1024, 1, B), 256, 0, stream>>>(
        gFT, gP, C2, W, (long)W * C2, (long)C2 * W4);

    // 4) logits = thetaT x phi  -> f32 [W][W4]
    wmma_gemm<1, 0, false, false, kUseTdm><<<dim3(W4 / 128, W / 64, B), 256, 0, stream>>>(
        thetaT, phiPT, logits, W, W4, C8, C8, W4,
        (long)W * C8, (long)W4 * C8, (long)W * W4, nullptr, nullptr, 0, nullptr);

    // 5) softmax over v -> f16 attn [W][W4]
    softmax512<<<(B * W) / 8, 256, 0, stream>>>(logits, attnH, B * W);

    // 6) attn_g = gP x attn^T -> attngT [W][C2]
    wmma_gemm<1, 0, true, true, kUseTdm><<<dim3(W / 128, C2 / 64, B), 256, 0, stream>>>(
        gP, attnH, attngT, C2, W, W4, W4, C2,
        (long)C2 * W4, (long)W * W4, (long)W * C2, nullptr, nullptr, 0, nullptr);

    // 7) x = 2*src + gamma * (W_o x attn_g) -> xT [W][CH] f16
    wmma_gemm<1, 1, true, true, kUseTdm><<<dim3(W / 128, CH / 64, B), 256, 0, stream>>>(
        woH, attngT, xT, CH, W, C2, C2, CH,
        0, (long)W * C2, (long)W * CH, nullptr, src, (long)CH * W, gamma);

    // 8) conv FFN (taps = row-shift on transposed activations; vector path)
    wmma_gemm<3, 2, true, true, false><<<dim3(W / 128, C2 / 64, B), 256, 0, stream>>>(
        wc1p, xT, h1T, C2, W, CH, CH, C2,
        0, (long)W * CH, (long)W * C2, b_c1, nullptr, 0, nullptr);
    wmma_gemm<3, 2, true, true, false><<<dim3(W / 128, C2 / 64, B), 256, 0, stream>>>(
        wc2p, h1T, h2T, C2, W, C2, C2, C2,
        0, (long)W * C2, (long)W * C2, b_c2, nullptr, 0, nullptr);
    wmma_gemm<3, 2, true, true, false><<<dim3(W / 128, C2 / 64, B), 256, 0, stream>>>(
        wc3p, h2T, h3T, C2, W, C2, C2, C2,
        0, (long)W * C2, (long)W * C2, b_c3, nullptr, 0, nullptr);
    wmma_gemm<3, 3, false, false, false><<<dim3(W / 128, CH / 64, B), 256, 0, stream>>>(
        wc4p, h3T, out, CH, W, C2, C2, W,
        0, (long)W * C2, (long)CH * W, b_c4, src, (long)CH * W, nullptr);
}
